// GNNLayer_16707422781816
// MI455X (gfx1250) — matrix operand order repaired and checked by reading, not compiled, not run on hardware
//
#include <hip/hip_runtime.h>

typedef __bf16 bf16;
typedef bf16  v16bf  __attribute__((ext_vector_type(16)));
typedef bf16  bf16x2 __attribute__((ext_vector_type(2)));
typedef float v8f    __attribute__((ext_vector_type(8)));

#define DD 512
#define BB 4
#define NN 20000
#define ROWS (BB * NN)            /* 80000 flattened (b,n) rows */
#define W_ELEMS (DD * DD)         /* 262144 */
#define AGG_BYTES ((size_t)ROWS * DD * sizeof(float))
#define LN_EPS 1e-5f

// ---------------- kernel 0: zero the aggregation buffer ----------------
__global__ void zero_agg(float4* __restrict__ agg) {
    size_t i = (size_t)blockIdx.x * blockDim.x + threadIdx.x;
    agg[i] = make_float4(0.f, 0.f, 0.f, 0.f);
}

// ---------------- kernel 1: split W into bf16 hi/lo --------------------
__global__ void conv_w(const float* __restrict__ W,
                       bf16* __restrict__ wh, bf16* __restrict__ wl) {
    int i = blockIdx.x * 256 + threadIdx.x;
    float w = W[i];
    bf16 h = (bf16)w;
    wh[i] = h;
    wl[i] = (bf16)(w - (float)h);
}

// ---------------- kernel 2: edge scatter-add ----------------------------
// idx = ((e*BB + b) * 128 + c) ; each thread moves 4 consecutive floats.
__global__ void scatter_add(const float* __restrict__ x,
                            const int* __restrict__ edges,
                            float* __restrict__ agg, int E) {
    long long idx = (long long)blockIdx.x * 256 + threadIdx.x;
    int c = (int)(idx & 127);
    long long t = idx >> 7;
    int b = (int)(t & 3);
    int e = (int)(t >> 2);
    if (e >= E) return;
    int src = edges[2 * e + 0];
    int dst = edges[2 * e + 1];
    const float4 v = *(const float4*)(x + ((size_t)b * NN + src) * DD + c * 4);
    float* o = agg + ((size_t)b * NN + dst) * DD + c * 4;
    atomicAdd(o + 0, v.x);
    atomicAdd(o + 1, v.y);
    atomicAdd(o + 2, v.z);
    atomicAdd(o + 3, v.w);
}

// ---------------- kernel 3: fused GEMM + bias + LN + GELU + residual ----
// 256 threads = 8 wave32. Block handles 16 rows x 512 cols.
// Wave w owns columns [64w, 64w+64): 4 C tiles of 16x16 (v8f each).
// Split-bf16: y = Ahi*Bhi + Ahi*Blo + Alo*Bhi (f32 accumulate).
__global__ __launch_bounds__(256)
void gemm_ln_gelu(const float* __restrict__ agg,
                  const bf16* __restrict__ wh, const bf16* __restrict__ wl,
                  const float* __restrict__ bias,
                  const float* __restrict__ gamma, const float* __restrict__ beta,
                  const float* __restrict__ x, float* __restrict__ out) {
    const int tid = threadIdx.x;
    const int wave = tid >> 5;
    const int lane = tid & 31;
    const int half = lane >> 4;      // 0: lanes 0-15, 1: lanes 16-31
    const int lane16 = lane & 15;
    const int rowBase = blockIdx.x * 16;
    const int colBase = wave * 64;

    __shared__ float s_sum[16];
    __shared__ float s_sq[16];
    if (tid < 16) { s_sum[tid] = 0.f; s_sq[tid] = 0.f; }
    __syncthreads();

    const float* aggRow = agg + (size_t)(rowBase + lane16) * DD;

    v8f c[4];
#pragma unroll
    for (int t = 0; t < 4; ++t)
#pragma unroll
        for (int r = 0; r < 8; ++r) c[t][r] = 0.f;

    for (int kk = 0; kk < DD; kk += 32) {
        // ---- A tile 16x32 bf16, ISA layout: VGPR v holds K pair at
        //      k0 = 2v + (v>=4 ? 8 : 0) + 8*half; lane16 = row M.
        v16bf a_hi, a_lo;
#pragma unroll
        for (int v = 0; v < 8; ++v) {
            const int k0 = 2 * v + ((v >= 4) ? 8 : 0) + 8 * half;
            const float2 av = *(const float2*)(aggRow + kk + k0);
            const bf16 h0 = (bf16)av.x;
            const bf16 h1 = (bf16)av.y;
            a_hi[2 * v]     = h0;
            a_hi[2 * v + 1] = h1;
            a_lo[2 * v]     = (bf16)(av.x - (float)h0);
            a_lo[2 * v + 1] = (bf16)(av.y - (float)h1);
        }
#pragma unroll
        for (int t = 0; t < 4; ++t) {
            const int col = colBase + t * 16 + lane16;   // output feature o
            const bf16* whp = wh + (size_t)col * DD + kk;
            const bf16* wlp = wl + (size_t)col * DD + kk;
            // ---- B tile 32x16 bf16: lane16 = col N, VGPR v holds K pair at
            //      k0 = 2v + (v>=4 ? 8 : 0) + 8*half (same order as A).
            v16bf b_hi, b_lo;
#pragma unroll
            for (int v = 0; v < 8; ++v) {
                const int k0 = 2 * v + ((v >= 4) ? 8 : 0) + 8 * half;
                const bf16x2 bh = *(const bf16x2*)(whp + k0);
                const bf16x2 bl = *(const bf16x2*)(wlp + k0);
                b_hi[2 * v]     = bh.x;
                b_hi[2 * v + 1] = bh.y;
                b_lo[2 * v]     = bl.x;
                b_lo[2 * v + 1] = bl.y;
            }
            c[t] = __builtin_amdgcn_wmma_f32_16x16x32_bf16(
                false, a_hi, false, b_hi, (short)0, c[t], false, false);
            c[t] = __builtin_amdgcn_wmma_f32_16x16x32_bf16(
                false, a_hi, false, b_lo, (short)0, c[t], false, false);
            c[t] = __builtin_amdgcn_wmma_f32_16x16x32_bf16(
                false, a_lo, false, b_hi, (short)0, c[t], false, false);
        }
    }

    // ---- bias + per-row sum / sumsq (C layout: VGPR r -> row r + 8*half,
    //      lane16 -> col within tile).
    float p[8], q[8];
#pragma unroll
    for (int r = 0; r < 8; ++r) { p[r] = 0.f; q[r] = 0.f; }
#pragma unroll
    for (int t = 0; t < 4; ++t) {
        const float bv = bias[colBase + t * 16 + lane16];
#pragma unroll
        for (int r = 0; r < 8; ++r) {
            const float y = c[t][r] + bv;
            c[t][r] = y;
            p[r] += y;
            q[r] += y * y;
        }
    }
    // butterfly over the 16-lane half (rows live separately per half)
#pragma unroll
    for (int r = 0; r < 8; ++r) {
#pragma unroll
        for (int m = 8; m >= 1; m >>= 1) {
            p[r] += __shfl_xor(p[r], m, 32);
            q[r] += __shfl_xor(q[r], m, 32);
        }
    }
    if (lane16 == 0) {
#pragma unroll
        for (int r = 0; r < 8; ++r) {
            atomicAdd(&s_sum[r + 8 * half], p[r]);   // ds_add_f32
            atomicAdd(&s_sq[r + 8 * half], q[r]);
        }
    }
    __syncthreads();

    float mean[8], inv[8];
#pragma unroll
    for (int r = 0; r < 8; ++r) {
        const int m = r + 8 * half;
        const float mu = s_sum[m] * (1.f / 512.f);
        mean[r] = mu;
        inv[r] = rsqrtf(s_sq[m] * (1.f / 512.f) - mu * mu + LN_EPS);
    }

    // ---- LN * gamma + beta, exact GELU, + residual, store
#pragma unroll
    for (int t = 0; t < 4; ++t) {
        const int col = colBase + t * 16 + lane16;
        const float g  = gamma[col];
        const float be = beta[col];
#pragma unroll
        for (int r = 0; r < 8; ++r) {
            const size_t off = (size_t)(rowBase + r + 8 * half) * DD + col;
            const float yn = (c[t][r] - mean[r]) * inv[r] * g + be;
            const float gl = 0.5f * yn * (1.f + erff(yn * 0.70710678118654752f));
            out[off] = gl + x[off];
        }
    }
}

// ---------------------------------------------------------------------------
extern "C" void kernel_launch(void* const* d_in, const int* in_sizes, int n_in,
                              void* d_out, int out_size, void* d_ws, size_t ws_size,
                              hipStream_t stream) {
    const float* x     = (const float*)d_in[0];
    const int*   edges = (const int*)d_in[1];
    const float* W     = (const float*)d_in[2];
    const float* b     = (const float*)d_in[3];
    const float* gamma = (const float*)d_in[4];
    const float* beta  = (const float*)d_in[5];
    float* out = (float*)d_out;
    const int E = in_sizes[1] / 2;

    float* agg = (float*)d_ws;
    bf16*  wh  = (bf16*)((char*)d_ws + AGG_BYTES);
    bf16*  wl  = wh + W_ELEMS;

    // 1) zero agg (40.96M floats as float4)
    zero_agg<<<(ROWS * DD / 4) / 256, 256, 0, stream>>>((float4*)agg);
    // 2) split W into bf16 hi/lo
    conv_w<<<W_ELEMS / 256, 256, 0, stream>>>(W, wh, wl);
    // 3) scatter-add aggregation
    const long long total = (long long)E * BB * (DD / 4);
    scatter_add<<<(int)((total + 255) / 256), 256, 0, stream>>>(x, edges, agg, E);
    // 4) fused GEMM + LN + GELU + residual
    gemm_ln_gelu<<<ROWS / 16, 256, 0, stream>>>(agg, wh, wl, b, gamma, beta, x, out);
}
